// CustomAttention_23252952940655
// MI455X (gfx1250) — matrix operand order repaired and checked by reading, not compiled, not run on hardware
//
#include <hip/hip_runtime.h>
#include <hip/hip_fp16.h>
#include <hip/hip_bf16.h>
#include <cstdint>

typedef __attribute__((ext_vector_type(16))) _Float16 v16h;
typedef __attribute__((ext_vector_type(8)))  float    v8f;

#define D_MODEL   1024
#define N_HEADS   16
#define HEAD_DIM  64
#define BATCH     4
#define SEQ       2048
#define M_TOKENS  (BATCH * SEQ)

// ---------------------------------------------------------------------------
// WMMA helpers (CDNA5 gfx1250, wave32). Operand layouts used by this file:
//  A (16x32 f16): lane L holds row L&15; halves = K (L<16?0:16)+0..15 (contig),
//                 same k order as B (not AMD's table order).
//  B (32x16 f16): lane L holds col L&15; halves = K (L<16?0:16)+0..15 (contig).
//  C/D (16x16 f32): lane L holds col L&15; VGPR v = row v + (L<16?0:8).
// ---------------------------------------------------------------------------
static __device__ inline v8f wmma_f16(v16h a, v16h b, v8f c) {
  return __builtin_amdgcn_wmma_f32_16x16x32_f16(
      /*neg_a=*/false, a, /*neg_b=*/false, b,
      /*c_mod=*/(short)0, c, /*reuse_a=*/false, /*reuse_b=*/false);
}

// rowp points at A[row, k0]; klo = (lane<16)?0:8
static __device__ inline v16h pack_a_f32(const float* __restrict__ rowp, int klo) {
  v16h a;
  const float* p = rowp + 2 * klo;
#pragma unroll
  for (int i = 0; i < 8; ++i) { a[i] = (_Float16)p[i]; a[i + 8] = (_Float16)p[i + 8]; }
  return a;
}
static __device__ inline v16h pack_a_f16(const _Float16* __restrict__ rowp, int klo) {
  v16h a;
  const _Float16* p = rowp + 2 * klo;
#pragma unroll
  for (int i = 0; i < 8; ++i) { a[i] = p[i]; a[i + 8] = p[i + 8]; }
  return a;
}

// ---------------------------------------------------------------------------
// CDNA5 async global->LDS copy (16B per lane), tracked by ASYNCcnt.
// LDS destination = low 32 bits of the generic shared pointer (flat LDS
// addresses are {shared_aperture, lds_offset}; hardware adds LDS_BASE).
// ---------------------------------------------------------------------------
static __device__ inline void async_copy_b128(void* lds_dst, const void* gsrc) {
  uint32_t loff = (uint32_t)(uintptr_t)lds_dst;
  unsigned long long ga = (unsigned long long)(uintptr_t)gsrc;
  asm volatile("global_load_async_to_lds_b128 %0, %1, off"
               :: "v"(loff), "v"(ga) : "memory");
}
#define WAIT_ASYNCCNT(N) asm volatile("s_wait_asynccnt " #N ::: "memory")

// ---------------------------------------------------------------------------
// fp32 -> f16 weight conversion
// ---------------------------------------------------------------------------
__global__ void cvt_f32_f16(const float* __restrict__ src,
                            _Float16* __restrict__ dst, int n) {
  int i = blockIdx.x * blockDim.x + threadIdx.x;
  if (i < n) dst[i] = (_Float16)src[i];
}

// ---------------------------------------------------------------------------
// Projection GEMM: Out[i,j] = f16( sum_d X[i,d]*W[j,d] + bias[j] )
// One wave -> 32x64 output tile (2 m-tiles x 4 n-tiles = 8 WMMAs / k-step).
// transposed==1 stores V as Vt[(b*1024 + j)*SEQ + s]  (for P@V WMMA operand).
// ---------------------------------------------------------------------------
__global__ __launch_bounds__(256) void proj_gemm(const float* __restrict__ X,
                                                 const _Float16* __restrict__ Wh,
                                                 const float* __restrict__ bias,
                                                 _Float16* __restrict__ Out,
                                                 int transposed) {
  const int wave = (blockIdx.x * blockDim.x + threadIdx.x) >> 5;
  const int lane = threadIdx.x & 31;
  const int ng = wave & 15;          // 1024/64 n-groups
  const int mg = wave >> 4;          // 8192/32 m-groups
  const int m0 = mg * 32, n0 = ng * 64;

  const int arow = lane & 15;
  const int klo  = (lane < 16) ? 0 : 8;
  const int bk   = (lane < 16) ? 0 : 16;
  const int n    = lane & 15;
  const int moff = (lane < 16) ? 0 : 8;

  const float*    Ar0 = X  + (size_t)(m0 + arow) * D_MODEL;
  const float*    Ar1 = Ar0 + (size_t)16 * D_MODEL;
  const _Float16* Bc  = Wh + (size_t)(n0 + n) * D_MODEL;   // +t*16*D_MODEL per n-tile

  v8f acc[2][4] = {};
#pragma unroll 2
  for (int k0 = 0; k0 < D_MODEL; k0 += 32) {
    v16h a0 = pack_a_f32(Ar0 + k0, klo);
    v16h a1 = pack_a_f32(Ar1 + k0, klo);
#pragma unroll
    for (int t = 0; t < 4; ++t) {
      v16h bm = *(const v16h*)(Bc + (size_t)t * 16 * D_MODEL + k0 + bk);
      acc[0][t] = wmma_f16(a0, bm, acc[0][t]);
      acc[1][t] = wmma_f16(a1, bm, acc[1][t]);
    }
  }

#pragma unroll
  for (int t = 0; t < 4; ++t) {
    const int col = n0 + t * 16 + n;
    const float bb = bias[col];
#pragma unroll
    for (int mt = 0; mt < 2; ++mt) {
#pragma unroll
      for (int v = 0; v < 8; ++v) {
        float val = acc[mt][t][v] + bb;
        int row = m0 + mt * 16 + v + moff;            // global token index
        if (!transposed) {
          Out[(size_t)row * D_MODEL + col] = (_Float16)val;
        } else {
          int bidx = row >> 11, s = row & (SEQ - 1);
          Out[((size_t)(bidx * D_MODEL + col)) * SEQ + s] = (_Float16)val;
        }
      }
    }
  }
}

// ---------------------------------------------------------------------------
// Flash attention. Block = 8 waves = 128 consecutive queries of ONE (b,h).
// K (32x64 f16) and transposed V (64x32 f16) tiles are DOUBLE-BUFFERED in LDS
// and filled with global_load_async_to_lds_b128 (ASYNCcnt): the next key
// block's DMA overlaps the current block's 8 WMMAs + softmax.
// scale = ln(1024)/sqrt(64); masks applied as -1e30.
// ---------------------------------------------------------------------------
__global__ __launch_bounds__(256) void flash_attn(const _Float16* __restrict__ Qh,
                                                  const _Float16* __restrict__ Kh,
                                                  const _Float16* __restrict__ Vt,
                                                  const int* __restrict__ mask,
                                                  const int* __restrict__ kpm,
                                                  _Float16* __restrict__ Aout) {
  __shared__ _Float16 Kl[2][32 * 64];    // [buf][key][dim]   8 KB
  __shared__ _Float16 Vl[2][64 * 32];    // [buf][dim][key]   8 KB
  __shared__ _Float16 Pl[8][16 * 32];    // per-wave P staging, 8 KB
  const int tid  = threadIdx.x;
  const int wid  = tid >> 5;
  const int lane = tid & 31;
  const int qc = blockIdx.x & 15;              // 16 query chunks of 128
  const int h  = (blockIdx.x >> 4) & 15;
  const int b  = blockIdx.x >> 8;
  const int q0 = qc * 128 + wid * 16;

  const int n    = lane & 15;
  const int half = (lane < 16) ? 0 : 1;
  const int klo  = half ? 8 : 0;
  const int moff = half ? 8 : 0;
  const float scale = 0.8664339756999316f;     // ln(1024)/8

  // cooperative staging indices (256 threads, 16B each)
  const int skey = tid >> 3, sch = (tid & 7) * 8;   // K: 32 keys x 64 dims
  const int sdim = tid >> 2, svh = (tid & 3) * 8;   // V: 64 dims x 32 keys
  const _Float16* Kg = Kh + (size_t)(b * SEQ + skey) * D_MODEL + h * HEAD_DIM + sch;
  const _Float16* Vg = Vt + (size_t)(b * D_MODEL + h * HEAD_DIM + sdim) * SEQ + svh;

  // Q tile: rows q0..q0+15, head dims 0..63 -> two A operands (kept in regs)
  const _Float16* qrowp = Qh + (size_t)(b * SEQ + q0 + n) * D_MODEL + h * HEAD_DIM;
  const v16h aq0 = pack_a_f16(qrowp, klo);        // d 0..31
  const v16h aq1 = pack_a_f16(qrowp + 32, klo);   // d 32..63

  v8f o0 = {}, o1 = {}, o2 = {}, o3 = {};
  float mrun[8], lrun[8];
#pragma unroll
  for (int v = 0; v < 8; ++v) { mrun[v] = -1e30f; lrun[v] = 0.0f; }

  const int* mrow = mask + (size_t)b * SEQ * SEQ;
  const int* kpmb = kpm + b * SEQ;

  const int NKB = SEQ / 32;   // 64 key blocks

  // prefetch tile 0 into buffer 0
  async_copy_b128(&Kl[0][skey * 64 + sch], Kg);
  async_copy_b128(&Vl[0][sdim * 32 + svh], Vg);

  for (int kb = 0; kb < NKB; ++kb) {
    const int cur = kb & 1;
    const int kbase = kb * 32;

    if (kb + 1 < NKB) {
      __syncthreads();   // everyone finished reading buffer cur^1 (iter kb-1)
      const int nb = kbase + 32;
      async_copy_b128(&Kl[cur ^ 1][skey * 64 + sch], Kg + (size_t)nb * D_MODEL);
      async_copy_b128(&Vl[cur ^ 1][sdim * 32 + svh], Vg + nb);
      WAIT_ASYNCCNT(2);  // tile kb landed (async loads complete in order)
    } else {
      WAIT_ASYNCCNT(0);
    }
    __syncthreads();     // tile kb visible block-wide

    const _Float16* kcur = &Kl[cur][0];
    const _Float16* vcur = &Vl[cur][0];

    // ---- E = (Q @ K^T) for a 16x32 block of keys (two 16x16 n-tiles) ----
    v8f e0 = {}, e1 = {};
    {
      const _Float16* kc0 = kcur + (0 * 16 + n) * 64;   // col = key, rows = dims
      const _Float16* kc1 = kcur + (1 * 16 + n) * 64;
      e0 = wmma_f16(aq0, *(const v16h*)(kc0 + half * 16),      e0);
      e0 = wmma_f16(aq1, *(const v16h*)(kc0 + 32 + half * 16), e0);
      e1 = wmma_f16(aq0, *(const v16h*)(kc1 + half * 16),      e1);
      e1 = wmma_f16(aq1, *(const v16h*)(kc1 + 32 + half * 16), e1);
    }

    // ---- scale + masks ----
#pragma unroll
    for (int v = 0; v < 8; ++v) {
      const int qrow = q0 + v + moff;
      const int k0i = kbase + n, k1i = kbase + 16 + n;
      bool ok0 = (mrow[(size_t)qrow * SEQ + k0i] != 0) && (kpmb[k0i] != 0);
      bool ok1 = (mrow[(size_t)qrow * SEQ + k1i] != 0) && (kpmb[k1i] != 0);
      e0[v] = ok0 ? e0[v] * scale : -1e30f;
      e1[v] = ok1 ? e1[v] * scale : -1e30f;
    }

    // ---- online softmax (row = v + 8*half, distributed over 16 lanes) ----
    float alpha[8];
#pragma unroll
    for (int v = 0; v < 8; ++v) {
      float t = fmaxf(e0[v], e1[v]);
      t = fmaxf(t, __shfl_xor(t, 1, 32));
      t = fmaxf(t, __shfl_xor(t, 2, 32));
      t = fmaxf(t, __shfl_xor(t, 4, 32));
      t = fmaxf(t, __shfl_xor(t, 8, 32));
      const float nm = fmaxf(mrun[v], t);
      alpha[v] = __expf(mrun[v] - nm);
      e0[v] = __expf(e0[v] - nm);
      e1[v] = __expf(e1[v] - nm);
      float rs = e0[v] + e1[v];
      rs += __shfl_xor(rs, 1, 32);
      rs += __shfl_xor(rs, 2, 32);
      rs += __shfl_xor(rs, 4, 32);
      rs += __shfl_xor(rs, 8, 32);
      lrun[v] = lrun[v] * alpha[v] + rs;
      mrun[v] = nm;
    }
#pragma unroll
    for (int v = 0; v < 8; ++v) {
      o0[v] *= alpha[v]; o1[v] *= alpha[v]; o2[v] *= alpha[v]; o3[v] *= alpha[v];
    }

    // ---- repack P (C-layout -> A-layout) through per-wave LDS ----
    _Float16* pl = Pl[wid];
#pragma unroll
    for (int v = 0; v < 8; ++v) {
      pl[(v + moff) * 32 + n]      = (_Float16)e0[v];
      pl[(v + moff) * 32 + 16 + n] = (_Float16)e1[v];
    }
    __asm__ volatile("s_wait_dscnt 0" ::: "memory");  // same-wave LDS RAW
    v16h ap;
    {
      const _Float16* prow = pl + n * 32;
#pragma unroll
      for (int i = 0; i < 8; ++i) { ap[i] = prow[2 * klo + i]; ap[i + 8] = prow[2 * klo + 8 + i]; }
    }

    // ---- O += P @ V  (B operand from LDS: col = head dim, contig keys) ----
    o0 = wmma_f16(ap, *(const v16h*)(vcur + (0 * 16 + n) * 32 + half * 16), o0);
    o1 = wmma_f16(ap, *(const v16h*)(vcur + (1 * 16 + n) * 32 + half * 16), o1);
    o2 = wmma_f16(ap, *(const v16h*)(vcur + (2 * 16 + n) * 32 + half * 16), o2);
    o3 = wmma_f16(ap, *(const v16h*)(vcur + (3 * 16 + n) * 32 + half * 16), o3);
  }

  // ---- normalize + store f16 attention output [token, feature] ----
#pragma unroll
  for (int v = 0; v < 8; ++v) {
    const float inv = 1.0f / lrun[v];
    const int row = b * SEQ + q0 + v + moff;
    _Float16* op = Aout + (size_t)row * D_MODEL + h * HEAD_DIM + n;
    op[0]  = (_Float16)(o0[v] * inv);
    op[16] = (_Float16)(o1[v] * inv);
    op[32] = (_Float16)(o2[v] * inv);
    op[48] = (_Float16)(o3[v] * inv);
  }
}

// ---------------------------------------------------------------------------
// Output projection: out[i,j] = sum_d Aout[i,d]*Wo[j,d] + bo[j]  (fp32 out)
// Same 32x64-per-wave tiling as proj_gemm, f16 A operand.
// ---------------------------------------------------------------------------
__global__ __launch_bounds__(256) void out_gemm(const _Float16* __restrict__ Ah,
                                                const _Float16* __restrict__ Wh,
                                                const float* __restrict__ bias,
                                                float* __restrict__ Out) {
  const int wave = (blockIdx.x * blockDim.x + threadIdx.x) >> 5;
  const int lane = threadIdx.x & 31;
  const int ng = wave & 15;
  const int mg = wave >> 4;
  const int m0 = mg * 32, n0 = ng * 64;

  const int arow = lane & 15;
  const int klo  = (lane < 16) ? 0 : 8;
  const int bk   = (lane < 16) ? 0 : 16;
  const int n    = lane & 15;
  const int moff = (lane < 16) ? 0 : 8;

  const _Float16* Ar0 = Ah + (size_t)(m0 + arow) * D_MODEL;
  const _Float16* Ar1 = Ar0 + (size_t)16 * D_MODEL;
  const _Float16* Bc  = Wh + (size_t)(n0 + n) * D_MODEL;

  v8f acc[2][4] = {};
#pragma unroll 2
  for (int k0 = 0; k0 < D_MODEL; k0 += 32) {
    v16h a0 = pack_a_f16(Ar0 + k0, klo);
    v16h a1 = pack_a_f16(Ar1 + k0, klo);
#pragma unroll
    for (int t = 0; t < 4; ++t) {
      v16h bm = *(const v16h*)(Bc + (size_t)t * 16 * D_MODEL + k0 + bk);
      acc[0][t] = wmma_f16(a0, bm, acc[0][t]);
      acc[1][t] = wmma_f16(a1, bm, acc[1][t]);
    }
  }

#pragma unroll
  for (int t = 0; t < 4; ++t) {
    const int col = n0 + t * 16 + n;
    const float bb = bias[col];
#pragma unroll
    for (int mt = 0; mt < 2; ++mt)
#pragma unroll
      for (int v = 0; v < 8; ++v)
        Out[(size_t)(m0 + mt * 16 + v + moff) * D_MODEL + col] = acc[mt][t][v] + bb;
  }
}

// ---------------------------------------------------------------------------
extern "C" void kernel_launch(void* const* d_in, const int* in_sizes, int n_in,
                              void* d_out, int out_size, void* d_ws, size_t ws_size,
                              hipStream_t stream) {
  const float* query = (const float*)d_in[0];
  const float* key   = (const float*)d_in[1];
  const float* value = (const float*)d_in[2];
  const int*   mask  = (const int*)d_in[3];
  const int*   kpm   = (const int*)d_in[4];
  const float* Wq = (const float*)d_in[5];  const float* bq = (const float*)d_in[6];
  const float* Wk = (const float*)d_in[7];  const float* bk = (const float*)d_in[8];
  const float* Wv = (const float*)d_in[9];  const float* bv = (const float*)d_in[10];
  const float* Wo = (const float*)d_in[11]; const float* bo = (const float*)d_in[12];

  // Workspace layout (f16 elements): 4x 1M weights, 3x 8M activations, 8M attn out
  _Float16* ws   = (_Float16*)d_ws;
  const size_t MB1 = (size_t)1 << 20;   // 1M halfs
  _Float16* Wqh  = ws;
  _Float16* Wkh  = ws + 1 * MB1;
  _Float16* Wvh  = ws + 2 * MB1;
  _Float16* Woh  = ws + 3 * MB1;
  _Float16* Qh   = ws + 4 * MB1;        // [8192,1024]
  _Float16* Kh   = ws + 12 * MB1;       // [8192,1024]
  _Float16* Vt   = ws + 20 * MB1;       // [B*1024, 2048] transposed
  _Float16* Aout = ws + 28 * MB1;       // [8192,1024]

  const int WN = D_MODEL * D_MODEL;     // 1M
  cvt_f32_f16<<<WN / 256, 256, 0, stream>>>(Wq, Wqh, WN);
  cvt_f32_f16<<<WN / 256, 256, 0, stream>>>(Wk, Wkh, WN);
  cvt_f32_f16<<<WN / 256, 256, 0, stream>>>(Wv, Wvh, WN);
  cvt_f32_f16<<<WN / 256, 256, 0, stream>>>(Wo, Woh, WN);

  // (8192/32)*(1024/64) = 4096 wave-tiles; 8 waves/block -> 512 blocks
  const int gemm_blocks = (M_TOKENS / 32) * (D_MODEL / 64) / 8;
  proj_gemm<<<gemm_blocks, 256, 0, stream>>>(query, Wqh, bq, Qh, 0);
  proj_gemm<<<gemm_blocks, 256, 0, stream>>>(key,   Wkh, bk, Kh, 0);
  proj_gemm<<<gemm_blocks, 256, 0, stream>>>(value, Wvh, bv, Vt, 1);

  // one block per (b, h, 128-query chunk): 4*16*16 = 1024 blocks
  flash_attn<<<BATCH * N_HEADS * (SEQ / 128), 256, 0, stream>>>(
      Qh, Kh, Vt, mask, kpm, Aout);

  out_gemm<<<gemm_blocks, 256, 0, stream>>>(Aout, Woh, bo, (float*)d_out);
}
